// CotLayer_78494822301921
// MI455X (gfx1250) — matrix operand order repaired and checked by reading, not compiled, not run on hardware
//
#include <hip/hip_runtime.h>
#include <stdint.h>

// ---------------------------------------------------------------------------
// CotLayer for MI455X (gfx1250): bf16 WMMA implicit-GEMM convs + fused epilogues
// v2: each wave owns 2 cout tiles (A-fragment reuse, 1.5 loads/WMMA),
//     unconditional speculative prefetch placed before the WMMAs.
// ---------------------------------------------------------------------------

#define EPSC 1e-5f

static __device__ __forceinline__ unsigned short f2bf(float f) {
  union { float f; unsigned u; } x; x.f = f;
  unsigned r = x.u + 0x7FFFu + ((x.u >> 16) & 1u);   // round-to-nearest-even
  return (unsigned short)(r >> 16);
}

typedef __attribute__((ext_vector_type(16))) __bf16 v16bf;
typedef __attribute__((ext_vector_type(8)))  float  v8f;

union Frag16 { v16bf v; uint4 q[2]; };
union Acc8   { v8f   v; float f[8]; };

// ---------------------------------------------------------------------------
// Pack conv weights (OIHW fp32) -> fragment-ready bf16: WB[k>>5][n][k&31],
// GEMM reduction index k = (kh*KW+kw)*Cin + cin.
// ---------------------------------------------------------------------------
__global__ void cot_pack_weights(const float* __restrict__ src,
                                 unsigned short* __restrict__ dst,
                                 int Cout, int Cin, int KK) {
  long total = (long)Cout * Cin * KK;
  long stride = (long)gridDim.x * blockDim.x;
  for (long e = (long)blockIdx.x * blockDim.x + threadIdx.x; e < total; e += stride) {
    int n   = (int)(e / (Cin * KK));
    int r   = (int)(e - (long)n * Cin * KK);
    int ci  = r / KK;
    int kid = r - ci * KK;
    int k   = kid * Cin + ci;
    dst[((size_t)(k >> 5) * Cout + n) * 32 + (k & 31)] = f2bf(src[e]);
  }
}

// ---------------------------------------------------------------------------
// 3x3 conv (pad=1) as implicit GEMM, WMMA bf16 -> f32. One block computes a
// 16-pixel row segment x Cout; wave w owns cout tiles [32w,32w+32) and reuses
// one A fragment across two WMMAs. Optional channel-concat input.
// Fused: +bias, BatchNorm(eval), ReLU.
// ---------------------------------------------------------------------------
__global__ void cot_conv3x3_wmma(const float* __restrict__ src0,
                                 const float* __restrict__ src1, int c0,
                                 int Cin, int Cout,
                                 const unsigned short* __restrict__ wb,
                                 const float* __restrict__ bias,
                                 const float* __restrict__ bng,
                                 const float* __restrict__ bnb,
                                 const float* __restrict__ bnm,
                                 const float* __restrict__ bnv,
                                 int relu,
                                 float* __restrict__ dst) {
  extern __shared__ unsigned short sx[];  // [3][18][Cin] bf16, cin contiguous
  const int t  = blockIdx.x;              // 2048 tiles = 8 * 64 * 4
  const int b  = t >> 8;
  const int rr = t & 255;
  const int h  = rr >> 2;
  const int w0 = (rr & 3) << 4;
  const int tid = threadIdx.x, nthr = blockDim.x;
  const int c1 = Cin - c0;

  // ---- stage bf16 input patch (rows h-1..h+1, cols w0-1..w0+16) ----
  const int tot = 3 * 18 * Cin;
  for (int e = tid; e < tot; e += nthr) {
    int dh = e / (18 * Cin);
    int r2 = e - dh * 18 * Cin;
    int ci = r2 / 18;
    int j  = r2 - ci * 18;
    int hh = h + dh - 1;
    int wp = w0 + j - 1;
    float val = 0.f;
    if ((unsigned)hh < 64u && (unsigned)wp < 64u) {
      if (ci < c0) val = src0[(((size_t)b * c0 + ci) * 64 + hh) * 64 + wp];
      else         val = src1[(((size_t)b * c1 + (ci - c0)) * 64 + hh) * 64 + wp];
    }
    sx[(dh * 18 + j) * Cin + ci] = f2bf(val);
  }
  __syncthreads();

  const int wave = tid >> 5;
  const int lane = tid & 31;
  const int n0   = wave << 5;            // two 16-wide cout tiles per wave
  const int mlo  = lane & 15;
  const int hi   = lane >> 4;
  const int hi8  = hi << 3;

  Acc8 acc0, acc1;
#pragma unroll
  for (int i = 0; i < 8; ++i) { acc0.f[i] = 0.f; acc1.f[i] = 0.f; }

  int kt = 0;
  for (int kid = 0; kid < 9; ++kid) {
    const int kh = kid / 3, kw = kid - kh * 3;
    const int rowbase = (kh * 18 + mlo + kw) * Cin + (hi << 4);
    for (int cinb = 0; cinb < Cin; cinb += 32, ++kt) {
      Frag16 a, b0, b1;
      const unsigned short* ap = sx + rowbase + cinb;
      a.q[0] = *(const uint4*)(ap);
      a.q[1] = *(const uint4*)(ap + 8);
      const unsigned short* bp = wb + ((size_t)kt * Cout + n0 + mlo) * 32 + (hi << 4);
      b0.q[0] = *(const uint4*)(bp);
      b0.q[1] = *(const uint4*)(bp + 8);
      b1.q[0] = *(const uint4*)(bp + 512);       // cout tile +16
      b1.q[1] = *(const uint4*)(bp + 520);
      __builtin_prefetch(bp + (size_t)Cout * 32, 0, 1);  // speculative, next K tile
      acc0.v = __builtin_amdgcn_wmma_f32_16x16x32_bf16(
          false, a.v, false, b0.v, (short)0, acc0.v, false, false);
      acc1.v = __builtin_amdgcn_wmma_f32_16x16x32_bf16(
          false, a.v, false, b1.v, (short)0, acc1.v, false, false);
    }
  }

  // ---- fused bias + BN + ReLU epilogue (both cout tiles) ----
#pragma unroll
  for (int half = 0; half < 2; ++half) {
    const int n = n0 + (half << 4) + mlo;
    const Acc8& acc = half ? acc1 : acc0;
    float scale = 1.f, shift = 0.f;
    if (bng) {
      scale = bng[n] * rsqrtf(bnv[n] + EPSC);
      shift = bnb[n] - bnm[n] * scale;
    }
    const float bs = bias ? bias[n] : 0.f;
#pragma unroll
    for (int r = 0; r < 8; ++r) {
      int m = r + hi8;
      float y = (acc.f[r] + bs) * scale + shift;
      if (relu) y = fmaxf(y, 0.f);
      dst[(((size_t)b * Cout + n) * 64 + h) * 64 + (w0 + m)] = y;
    }
  }
}

// ---------------------------------------------------------------------------
// 1x1 conv as GEMM (same 2-cout-tile wave tiling), fused bias/BN/ReLU.
// ---------------------------------------------------------------------------
__global__ void cot_conv1x1_wmma(const float* __restrict__ src,
                                 int Cin, int Cout,
                                 const unsigned short* __restrict__ wb,
                                 const float* __restrict__ bias,
                                 const float* __restrict__ bng,
                                 const float* __restrict__ bnb,
                                 const float* __restrict__ bnm,
                                 const float* __restrict__ bnv,
                                 int relu,
                                 float* __restrict__ dst) {
  extern __shared__ unsigned short sx[];  // [16][Cin] bf16
  const int t  = blockIdx.x;
  const int b  = t >> 8;
  const int rr = t & 255;
  const int h  = rr >> 2;
  const int w0 = (rr & 3) << 4;
  const int tid = threadIdx.x, nthr = blockDim.x;

  const int tot = 16 * Cin;
  for (int e = tid; e < tot; e += nthr) {
    int m  = e & 15;
    int ci = e >> 4;
    float val = src[(((size_t)b * Cin + ci) * 64 + h) * 64 + (w0 + m)];
    sx[m * Cin + ci] = f2bf(val);
  }
  __syncthreads();

  const int wave = tid >> 5;
  const int lane = tid & 31;
  const int n0   = wave << 5;
  const int mlo  = lane & 15;
  const int hi   = lane >> 4;
  const int hi8  = hi << 3;

  Acc8 acc0, acc1;
#pragma unroll
  for (int i = 0; i < 8; ++i) { acc0.f[i] = 0.f; acc1.f[i] = 0.f; }

  const int ksteps = Cin >> 5;
  for (int kt = 0; kt < ksteps; ++kt) {
    Frag16 a, b0, b1;
    const unsigned short* ap = sx + mlo * Cin + (kt << 5) + (hi << 4);
    a.q[0] = *(const uint4*)(ap);
    a.q[1] = *(const uint4*)(ap + 8);
    const unsigned short* bp = wb + ((size_t)kt * Cout + n0 + mlo) * 32 + (hi << 4);
    b0.q[0] = *(const uint4*)(bp);
    b0.q[1] = *(const uint4*)(bp + 8);
    b1.q[0] = *(const uint4*)(bp + 512);
    b1.q[1] = *(const uint4*)(bp + 520);
    __builtin_prefetch(bp + (size_t)Cout * 32, 0, 1);
    acc0.v = __builtin_amdgcn_wmma_f32_16x16x32_bf16(
        false, a.v, false, b0.v, (short)0, acc0.v, false, false);
    acc1.v = __builtin_amdgcn_wmma_f32_16x16x32_bf16(
        false, a.v, false, b1.v, (short)0, acc1.v, false, false);
  }

#pragma unroll
  for (int half = 0; half < 2; ++half) {
    const int n = n0 + (half << 4) + mlo;
    const Acc8& acc = half ? acc1 : acc0;
    float scale = 1.f, shift = 0.f;
    if (bng) {
      scale = bng[n] * rsqrtf(bnv[n] + EPSC);
      shift = bnb[n] - bnm[n] * scale;
    }
    const float bs = bias ? bias[n] : 0.f;
#pragma unroll
    for (int r = 0; r < 8; ++r) {
      int m = r + hi8;
      float y = (acc.f[r] + bs) * scale + shift;
      if (relu) y = fmaxf(y, 0.f);
      dst[(((size_t)b * Cout + n) * 64 + h) * 64 + (w0 + m)] = y;
    }
  }
}

// ---------------------------------------------------------------------------
// GroupNorm over (9 channels, H, W) per (b, group). 256 blocks x 256 threads.
// ---------------------------------------------------------------------------
__global__ void cot_groupnorm(float* __restrict__ w3,
                              const float* __restrict__ gg,
                              const float* __restrict__ gb) {
  __shared__ float rs[256], rq[256];
  const int b = blockIdx.x >> 5, g = blockIdx.x & 31;
  float* base = w3 + ((size_t)b * 288 + g * 9) * 4096;
  const int N = 9 * 4096;
  float s = 0.f, q = 0.f;
  for (int e = threadIdx.x; e < N; e += blockDim.x) {
    float x = base[e]; s += x; q += x * x;
  }
  rs[threadIdx.x] = s; rq[threadIdx.x] = q;
  __syncthreads();
  for (int st = 128; st > 0; st >>= 1) {
    if ((int)threadIdx.x < st) {
      rs[threadIdx.x] += rs[threadIdx.x + st];
      rq[threadIdx.x] += rq[threadIdx.x + st];
    }
    __syncthreads();
  }
  const float mean = rs[0] / (float)N;
  const float var  = rq[0] / (float)N - mean * mean;
  const float inv  = rsqrtf(var + EPSC);
  for (int e = threadIdx.x; e < N; e += blockDim.x) {
    int ch = g * 9 + (e >> 12);
    base[e] = (base[e] - mean) * inv * gg[ch] + gb[ch];
  }
}

// ---------------------------------------------------------------------------
// Per-pixel dynamic 3x3 conv (weights shared by 8 consecutive channels),
// fused bn2 + swish.
// ---------------------------------------------------------------------------
__global__ void cot_localconv(const float* __restrict__ v,
                              const float* __restrict__ wd,
                              const float* __restrict__ g2,
                              const float* __restrict__ b2,
                              const float* __restrict__ m2,
                              const float* __restrict__ v2,
                              float* __restrict__ vdyn) {
  const long total = 8L * 256 * 4096;
  const long stride = (long)gridDim.x * blockDim.x;
  for (long idx = (long)blockIdx.x * blockDim.x + threadIdx.x; idx < total; idx += stride) {
    int w = (int)(idx & 63);
    int h = (int)((idx >> 6) & 63);
    int c = (int)((idx >> 12) & 255);
    int b = (int)(idx >> 20);
    const float* vb = v + ((idx >> 12) << 12);
    const float* wp = wd + (((size_t)b * 32 + (c >> 3)) * 9) * 4096 + (idx & 4095);
    float s = 0.f;
#pragma unroll
    for (int i = 0; i < 3; ++i)
#pragma unroll
      for (int j = 0; j < 3; ++j) {
        int hh = h + i - 1, wwp = w + j - 1;
        float xv = ((unsigned)hh < 64u && (unsigned)wwp < 64u) ? vb[hh * 64 + wwp] : 0.f;
        s += xv * wp[(i * 3 + j) * 4096];
      }
    float sc = g2[c] * rsqrtf(v2[c] + EPSC);
    float y  = s * sc + (b2[c] - m2[c] * sc);
    vdyn[idx] = y / (1.f + __expf(-y));  // swish
  }
}

// ---------------------------------------------------------------------------
// GAP: gap[b,c] = mean_{hw}(vdyn + k). Block per (b,c).
// ---------------------------------------------------------------------------
__global__ void cot_gap(const float* __restrict__ vdyn,
                        const float* __restrict__ kbuf,
                        float* __restrict__ gap) {
  __shared__ float rs[256];
  const int bc = blockIdx.x;
  const float* p1 = vdyn + (size_t)bc * 4096;
  const float* p2 = kbuf + (size_t)bc * 4096;
  float s = 0.f;
  for (int e = threadIdx.x; e < 4096; e += blockDim.x) s += p1[e] + p2[e];
  rs[threadIdx.x] = s;
  __syncthreads();
  for (int st = 128; st > 0; st >>= 1) {
    if ((int)threadIdx.x < st) rs[threadIdx.x] += rs[threadIdx.x + st];
    __syncthreads();
  }
  if (threadIdx.x == 0) gap[bc] = rs[0] * (1.f / 4096.f);
}

// ---------------------------------------------------------------------------
// SE MLP + radix-2 softmax. 8 blocks x 128 threads.
// ---------------------------------------------------------------------------
__global__ void cot_se(const float* __restrict__ gap,
                       const float* __restrict__ w1, const float* __restrict__ b1,
                       const float* __restrict__ sg, const float* __restrict__ sb,
                       const float* __restrict__ sm, const float* __restrict__ sv,
                       const float* __restrict__ w2, const float* __restrict__ b2,
                       float* __restrict__ att) {
  __shared__ float gsh[256], a1[128], a2[512];
  const int b = blockIdx.x, t = threadIdx.x;
  gsh[t] = gap[b * 256 + t];
  gsh[t + 128] = gap[b * 256 + t + 128];
  __syncthreads();
  float acc = b1[t];
  for (int c = 0; c < 256; ++c) acc += w1[t * 256 + c] * gsh[c];
  float sc = sg[t] * rsqrtf(sv[t] + EPSC);
  acc = acc * sc + (sb[t] - sm[t] * sc);
  a1[t] = fmaxf(acc, 0.f);
  __syncthreads();
  for (int o = t; o < 512; o += 128) {
    float s = b2[o];
    for (int c = 0; c < 128; ++c) s += w2[o * 128 + c] * a1[c];
    a2[o] = s;
  }
  __syncthreads();
  for (int c = t; c < 256; c += 128) {
    float e0 = a2[2 * c], e1 = a2[2 * c + 1];
    float mx = fmaxf(e0, e1);
    float x0 = __expf(e0 - mx), x1 = __expf(e1 - mx);
    float inv = 1.f / (x0 + x1);
    att[(b * 256 + c) * 2 + 0] = x0 * inv;
    att[(b * 256 + c) * 2 + 1] = x1 * inv;
  }
}

// ---------------------------------------------------------------------------
// Final blend: out = vdyn*att0 + k*att1.
// ---------------------------------------------------------------------------
__global__ void cot_blend(const float* __restrict__ vdyn,
                          const float* __restrict__ kbuf,
                          const float* __restrict__ att,
                          float* __restrict__ out) {
  const long total = 8L * 256 * 4096;
  const long stride = (long)gridDim.x * blockDim.x;
  for (long idx = (long)blockIdx.x * blockDim.x + threadIdx.x; idx < total; idx += stride) {
    long bc = idx >> 12;
    float a0 = att[bc * 2], a1 = att[bc * 2 + 1];
    out[idx] = vdyn[idx] * a0 + kbuf[idx] * a1;
  }
}

// ---------------------------------------------------------------------------
// Host launcher
// ---------------------------------------------------------------------------
extern "C" void kernel_launch(void* const* d_in, const int* in_sizes, int n_in,
                              void* d_out, int out_size, void* d_ws, size_t ws_size,
                              hipStream_t stream) {
  (void)in_sizes; (void)n_in; (void)out_size; (void)ws_size;

  // setup_inputs() flat order: x, then params dict depth-first.
  const float* x      = (const float*)d_in[0];
  const float* ke_w   = (const float*)d_in[1];
  const float* ke_g   = (const float*)d_in[2];
  const float* ke_b   = (const float*)d_in[3];
  const float* ke_m   = (const float*)d_in[4];
  const float* ke_v   = (const float*)d_in[5];
  const float* e1_w   = (const float*)d_in[6];
  const float* e1_bs  = (const float*)d_in[7];
  const float* e1_g   = (const float*)d_in[8];
  const float* e1_b   = (const float*)d_in[9];
  const float* e1_m   = (const float*)d_in[10];
  const float* e1_v   = (const float*)d_in[11];
  const float* e2_w   = (const float*)d_in[12];
  const float* e2_g   = (const float*)d_in[13];
  const float* e2_b   = (const float*)d_in[14];
  const float* e2_m   = (const float*)d_in[15];
  const float* e2_v   = (const float*)d_in[16];
  const float* e3_w   = (const float*)d_in[17];
  const float* e3_bs  = (const float*)d_in[18];
  const float* gn_g   = (const float*)d_in[19];
  const float* gn_b   = (const float*)d_in[20];
  const float* c11_w  = (const float*)d_in[21];
  const float* c11_g  = (const float*)d_in[22];
  const float* c11_b  = (const float*)d_in[23];
  const float* c11_m  = (const float*)d_in[24];
  const float* c11_v  = (const float*)d_in[25];
  const float* bn2_g  = (const float*)d_in[26];
  const float* bn2_b  = (const float*)d_in[27];
  const float* bn2_m  = (const float*)d_in[28];
  const float* bn2_v  = (const float*)d_in[29];
  const float* se_w1  = (const float*)d_in[30];
  const float* se_b1  = (const float*)d_in[31];
  const float* se_g   = (const float*)d_in[32];
  const float* se_b   = (const float*)d_in[33];
  const float* se_m   = (const float*)d_in[34];
  const float* se_v   = (const float*)d_in[35];
  const float* se_w2  = (const float*)d_in[36];
  const float* se_b2  = (const float*)d_in[37];

  char* ws = (char*)d_ws;
  size_t off = 0;
  auto alloc = [&](size_t bytes) -> void* {
    void* p = (void*)(ws + off);
    off += (bytes + 255) & ~(size_t)255;
    return p;
  };

  float* kbuf = (float*)alloc(8L * 256 * 4096 * 4);   // k embed
  float* w1   = (float*)alloc(8L * 256 * 4096 * 4);   // emb stage1
  float* w2   = (float*)alloc(8L * 128 * 4096 * 4);   // emb stage2
  float* w3   = (float*)alloc(8L * 288 * 4096 * 4);   // dyn weights
  float* vbuf = (float*)alloc(8L * 256 * 4096 * 4);   // value path
  float* vdyn = (float*)alloc(8L * 256 * 4096 * 4);   // dyn-conv out
  float* gapb = (float*)alloc(2048 * 4);
  float* attb = (float*)alloc(4096 * 4);
  unsigned short* wb_ke  = (unsigned short*)alloc(2304L * 256 * 2);
  unsigned short* wb_e1  = (unsigned short*)alloc(4608L * 256 * 2);
  unsigned short* wb_e2  = (unsigned short*)alloc(256L * 128 * 2);
  unsigned short* wb_e3  = (unsigned short*)alloc(128L * 288 * 2);
  unsigned short* wb_c11 = (unsigned short*)alloc(256L * 256 * 2);

  // --- weight packing (fp32 -> fragment-ready bf16) ---
  cot_pack_weights<<<256, 256, 0, stream>>>(ke_w,  wb_ke,  256, 256, 9);
  cot_pack_weights<<<256, 256, 0, stream>>>(e1_w,  wb_e1,  256, 512, 9);
  cot_pack_weights<<<64,  256, 0, stream>>>(e2_w,  wb_e2,  128, 256, 1);
  cot_pack_weights<<<64,  256, 0, stream>>>(e3_w,  wb_e3,  288, 128, 1);
  cot_pack_weights<<<64,  256, 0, stream>>>(c11_w, wb_c11, 256, 256, 1);

  // --- k = relu(bn(conv3x3(x))) ---  (8 waves = 256/32-cout tiles)
  cot_conv3x3_wmma<<<2048, 256, 3 * 18 * 256 * 2, stream>>>(
      x, x, 256, 256, 256, wb_ke, nullptr, ke_g, ke_b, ke_m, ke_v, 1, kbuf);
  // --- w1 = relu(bn(conv3x3(concat(x,k)) + b)) ---
  cot_conv3x3_wmma<<<2048, 256, 3 * 18 * 512 * 2, stream>>>(
      x, kbuf, 256, 512, 256, wb_e1, e1_bs, e1_g, e1_b, e1_m, e1_v, 1, w1);
  // --- w2 = relu(bn(conv1x1(w1))) ---  (4 waves, Cout=128)
  cot_conv1x1_wmma<<<2048, 128, 16 * 256 * 2, stream>>>(
      w1, 256, 128, wb_e2, nullptr, e2_g, e2_b, e2_m, e2_v, 1, w2);
  // --- w3 = conv1x1(w2) + b ---  (9 waves, Cout=288)
  cot_conv1x1_wmma<<<2048, 288, 16 * 128 * 2, stream>>>(
      w2, 128, 288, wb_e3, e3_bs, nullptr, nullptr, nullptr, nullptr, 0, w3);
  // --- groupnorm(w3, 32 groups) ---
  cot_groupnorm<<<256, 256, 0, stream>>>(w3, gn_g, gn_b);
  // --- v = bn(conv1x1(x)) ---
  cot_conv1x1_wmma<<<2048, 256, 16 * 256 * 2, stream>>>(
      x, 256, 256, wb_c11, nullptr, c11_g, c11_b, c11_m, c11_v, 0, vbuf);
  // --- vdyn = swish(bn2(local_conv(v, w3))) ---
  cot_localconv<<<16384, 256, 0, stream>>>(vbuf, w3, bn2_g, bn2_b, bn2_m, bn2_v, vdyn);
  // --- gap / SE / softmax ---
  cot_gap<<<2048, 256, 0, stream>>>(vdyn, kbuf, gapb);
  cot_se<<<8, 128, 0, stream>>>(gapb, se_w1, se_b1, se_g, se_b, se_m, se_v,
                                se_w2, se_b2, attb);
  // --- out = vdyn*a0 + k*a1 ---
  cot_blend<<<8192, 256, 0, stream>>>(vdyn, kbuf, attb, (float*)d_out);
}
